// GraphAutoEncoder_53386443489396
// MI455X (gfx1250) — compile-verified, matrix operand order checked by reading, not run on hardware
//
#include <hip/hip_runtime.h>

#define THREADS 256

typedef __attribute__((ext_vector_type(16))) __bf16 v16bf;
typedef __attribute__((ext_vector_type(8)))  __bf16 v8bf;
typedef __attribute__((ext_vector_type(8)))  float  v8f;

// ---------------------------------------------------------------------------
// Utility kernels
// ---------------------------------------------------------------------------
__global__ void zero_kernel(float* __restrict__ p, size_t n) {
  size_t i = blockIdx.x * (size_t)blockDim.x + threadIdx.x;
  size_t stride = gridDim.x * (size_t)blockDim.x;
  for (; i < n; i += stride) p[i] = 0.f;
}

// W [K x Nout] f32 row-major -> Wt [Nout x ldw] bf16 row-major, K zero-padded
// to a multiple of 32 so WMMA B-fragment loads never need K guards.
// Grid: (ceil(ldw/THREADS), Nout)
__global__ void transpose_cast_kernel(const float* __restrict__ W, __bf16* __restrict__ Wt,
                                      int K, int Nout, int ldw) {
  int k = blockIdx.x * THREADS + threadIdx.x;
  int n = blockIdx.y;
  if (k >= ldw) return;
  Wt[(size_t)n * ldw + k] = (k < K) ? (__bf16)W[(size_t)k * Nout + n] : (__bf16)0.f;
}

// agg[row] += val * S[col]; 4 features/thread, hardware f32 atomics.
// F is always a power of two here; lf4 = log2(F/4).
__global__ void spmm_kernel(const int* __restrict__ rows, const int* __restrict__ cols,
                            const float* __restrict__ vals, const float* __restrict__ S,
                            float* __restrict__ AGG, int E, int F, int lf4) {
  const int f4mask = (1 << lf4) - 1;
  size_t total = (size_t)E << lf4;
  size_t i = blockIdx.x * (size_t)blockDim.x + threadIdx.x;
  size_t stride = gridDim.x * (size_t)blockDim.x;
  for (; i < total; i += stride) {
    int e = (int)(i >> lf4);
    int j = ((int)i & f4mask) << 2;
    int c = cols[e];
    int r = rows[e];
    float v = vals[e];
    const float4 s = *(const float4*)(S + (size_t)c * F + j);
    float* dst = AGG + (size_t)r * F + j;
    unsafeAtomicAdd(dst + 0, v * s.x);
    unsafeAtomicAdd(dst + 1, v * s.y);
    unsafeAtomicAdd(dst + 2, v * s.z);
    unsafeAtomicAdd(dst + 3, v * s.w);
  }
}

// out = (in + bias) with optional ReLU (in-place safe). F is a power of two.
__global__ void bias_act_kernel(const float* in, const float* __restrict__ bias,
                                float* out, size_t total, int fmask, int relu) {
  size_t i = blockIdx.x * (size_t)blockDim.x + threadIdx.x;
  size_t stride = gridDim.x * (size_t)blockDim.x;
  for (; i < total; i += stride) {
    float v = in[i] + bias[(int)i & fmask];
    if (relu) v = fmaxf(v, 0.f);
    out[i] = v;
  }
}

// ---------------------------------------------------------------------------
// WMMA GEMM:  C[M x Nout] = A[M x K](f32 -> bf16 on the fly) @ Wt[Nout x ldw]^T
// Requires M % 16 == 0 (true here: 20000 = 1250*16). Branch-free inner loop:
//  - wave m0 clamped into range (duplicate tiles store identical values)
//  - B column clamped for loads; only the store checks n < Nout
//  - K tail handled by one extra iteration with clamped loads + value selects
//  - all loads of a K-step are issued before the first wmma so the scheduler
//    can use partial s_wait_loadcnt values and overlap L2 latency
// mode: 0 = plain, 1 = +bias, 2 = +bias+relu
// ---------------------------------------------------------------------------
template <int NT, bool ALIGNED>
__global__ void __launch_bounds__(THREADS) wmma_gemm_kernel(
    const float* __restrict__ A, const __bf16* __restrict__ Bt,
    const float* __restrict__ bias, float* __restrict__ C,
    int M, int K, int Nout, int lda, int ldw, int mode) {
  const int wave = threadIdx.x >> 5;
  const int lane = threadIdx.x & 31;
  const int half = lane >> 4;
  const int l16  = lane & 15;

  int m0 = blockIdx.x * 128 + wave * 16;
  if (m0 > M - 16) m0 = M - 16;              // M % 16 == 0: full-tile clamp
  const int n0 = blockIdx.y * (NT * 16);

  const float* pa = A + (size_t)(m0 + l16) * lda;  // this lane's A row

  // clamped B row pointers (per tile)
  const __bf16* pb[NT];
#pragma unroll
  for (int t = 0; t < NT; ++t) {
    int n = n0 + t * 16 + l16;
    int nc = (n < Nout) ? n : (Nout - 1);
    pb[t] = Bt + (size_t)nc * ldw;
  }

  v8f acc[NT] = {};

  const int Kmain = K & ~31;
  for (int k0 = 0; k0 < Kmain; k0 += 32) {
    // ISA 16-bit A-fragment layout: lanes 0-15: K=[k0..k0+7],[k0+16..k0+23];
    // lanes 16-31: K=[k0+8..k0+15],[k0+24..k0+31].
    const int klo = k0 + half * 8;
    const int khi = klo + 16;

    // ---- issue A loads ----
    float ar[16];
    if (ALIGNED) {
      float4 f0 = *(const float4*)(pa + klo);
      float4 f1 = *(const float4*)(pa + klo + 4);
      float4 g0 = *(const float4*)(pa + khi);
      float4 g1 = *(const float4*)(pa + khi + 4);
      ar[0] = f0.x;  ar[1] = f0.y;  ar[2] = f0.z;  ar[3] = f0.w;
      ar[4] = f1.x;  ar[5] = f1.y;  ar[6] = f1.z;  ar[7] = f1.w;
      ar[8] = g0.x;  ar[9] = g0.y;  ar[10] = g0.z; ar[11] = g0.w;
      ar[12] = g1.x; ar[13] = g1.y; ar[14] = g1.z; ar[15] = g1.w;
    } else {
#pragma unroll
      for (int t = 0; t < 8; ++t) ar[t] = pa[klo + t];
#pragma unroll
      for (int t = 0; t < 8; ++t) ar[8 + t] = pa[khi + t];
    }

    // ---- issue ALL B loads before any wmma ----
    v8bf blo[NT], bhi[NT];
#pragma unroll
    for (int t = 0; t < NT; ++t) {
      blo[t] = *(const v8bf*)(pb[t] + klo);
      bhi[t] = *(const v8bf*)(pb[t] + khi);
    }

    // prefetch the streaming A operand two K-steps ahead (global_prefetch_b8)
    __builtin_prefetch(pa + klo + 64, 0, 3);

    // ---- convert A while B loads are in flight ----
    v16bf a;
#pragma unroll
    for (int t = 0; t < 16; ++t) a[t] = (__bf16)ar[t];

    // ---- wmma chain ----
#pragma unroll
    for (int t = 0; t < NT; ++t) {
      v16bf b = __builtin_shufflevector(blo[t], bhi[t], 0, 1, 2, 3, 4, 5, 6, 7,
                                        8, 9, 10, 11, 12, 13, 14, 15);
      acc[t] = __builtin_amdgcn_wmma_f32_16x16x32_bf16(
          false, a, false, b, (short)0, acc[t], false, false);
    }
  }

  if (Kmain < K) {                           // K tail: clamped loads + selects
    const int klo = Kmain + half * 8;
    const int khi = klo + 16;
    v16bf a;
#pragma unroll
    for (int t = 0; t < 8; ++t) {
      int k = klo + t;
      int kc = (k < K) ? k : (K - 1);
      float v = pa[kc];
      a[t] = (k < K) ? (__bf16)v : (__bf16)0.f;
    }
#pragma unroll
    for (int t = 0; t < 8; ++t) {
      int k = khi + t;
      int kc = (k < K) ? k : (K - 1);
      float v = pa[kc];
      a[8 + t] = (k < K) ? (__bf16)v : (__bf16)0.f;
    }
    // Wt rows are zero-padded to ldw (multiple of 32): no K guard needed.
    v8bf blo[NT], bhi[NT];
#pragma unroll
    for (int t = 0; t < NT; ++t) {
      blo[t] = *(const v8bf*)(pb[t] + klo);
      bhi[t] = *(const v8bf*)(pb[t] + khi);
    }
#pragma unroll
    for (int t = 0; t < NT; ++t) {
      v16bf b = __builtin_shufflevector(blo[t], bhi[t], 0, 1, 2, 3, 4, 5, 6, 7,
                                        8, 9, 10, 11, 12, 13, 14, 15);
      acc[t] = __builtin_amdgcn_wmma_f32_16x16x32_bf16(
          false, a, false, b, (short)0, acc[t], false, false);
    }
  }

  // C/D layout: VGPR v of lane -> row m0 + v + 8*half, col n0 + t*16 + l16
#pragma unroll
  for (int t = 0; t < NT; ++t) {
    int n = n0 + t * 16 + l16;
    if (n >= Nout) continue;
    float bv = (mode != 0) ? bias[n] : 0.f;
#pragma unroll
    for (int v = 0; v < 8; ++v) {
      int row = m0 + v + half * 8;
      float val = acc[t][v] + bv;
      if (mode == 2) val = fmaxf(val, 0.f);
      C[(size_t)row * Nout + n] = val;
    }
  }
}

// ---------------------------------------------------------------------------
// Host orchestration
// ---------------------------------------------------------------------------
static inline void launch_gemm(const float* A, const __bf16* Bt, const float* bias,
                               float* C, int M, int K, int Nout, int lda, int ldw,
                               int mode, hipStream_t stream) {
  const bool aligned = ((lda & 3) == 0);
  if (Nout >= 128) {
    dim3 g((M + 127) / 128, (Nout + 127) / 128);
    if (aligned)
      wmma_gemm_kernel<8, true><<<g, THREADS, 0, stream>>>(A, Bt, bias, C, M, K, Nout, lda, ldw, mode);
    else
      wmma_gemm_kernel<8, false><<<g, THREADS, 0, stream>>>(A, Bt, bias, C, M, K, Nout, lda, ldw, mode);
  } else {
    dim3 g((M + 127) / 128, (Nout + 15) / 16);
    if (aligned)
      wmma_gemm_kernel<1, true><<<g, THREADS, 0, stream>>>(A, Bt, bias, C, M, K, Nout, lda, ldw, mode);
    else
      wmma_gemm_kernel<1, false><<<g, THREADS, 0, stream>>>(A, Bt, bias, C, M, K, Nout, lda, ldw, mode);
  }
}

static inline int ilog2(int x) { int l = 0; while ((1 << l) < x) ++l; return l; }

extern "C" void kernel_launch(void* const* d_in, const int* in_sizes, int n_in,
                              void* d_out, int out_size, void* d_ws, size_t ws_size,
                              hipStream_t stream) {
  (void)n_in; (void)out_size; (void)ws_size;
  const int IN_DIM = 3703, LATENT = 16;
  const int ENC[5] = {3703, 1024, 512, 256, 16};
  const int DEC[5] = {16, 256, 512, 1024, 3703};
  const int N = in_sizes[0] / IN_DIM;
  const int E = in_sizes[1];

  const float* x     = (const float*)d_in[0];
  const int*   erow  = (const int*)d_in[1];
  const int*   ecol  = (const int*)d_in[2];
  const float* eval_ = (const float*)d_in[3];

  const float* Wf[8];
  const float* Bf[8];
  int Kd[8], Nd[8], LDW[8];
  for (int i = 0; i < 8; ++i) {
    Wf[i] = (const float*)d_in[4 + 2 * i];
    Bf[i] = (const float*)d_in[5 + 2 * i];
    if (i < 4) { Kd[i] = ENC[i];     Nd[i] = ENC[i + 1]; }
    else       { Kd[i] = DEC[i - 4]; Nd[i] = DEC[i - 3]; }
    LDW[i] = (Kd[i] + 31) & ~31;
  }

  // workspace: 8 transposed bf16 weights (~18 MB) + two f32 ping-pong buffers
  char* ws = (char*)d_ws;
  size_t off = 0;
  auto take = [&](size_t bytes) -> char* {
    char* p = ws + off;
    off = (off + bytes + 255) & ~(size_t)255;
    return p;
  };
  __bf16* Wt[8];
  for (int i = 0; i < 8; ++i)
    Wt[i] = (__bf16*)take((size_t)Nd[i] * LDW[i] * sizeof(__bf16));
  float* bufA = (float*)take((size_t)N * 1024 * sizeof(float));
  float* bufB = (float*)take((size_t)N * 1024 * sizeof(float));

  auto blocks1d = [](size_t total) -> dim3 {
    size_t b = (total + THREADS - 1) / THREADS;
    if (b > 1048576) b = 1048576;
    return dim3((unsigned)b);
  };

  // per-call weight transpose+cast (deterministic; ~36 MB read total)
  for (int i = 0; i < 8; ++i) {
    dim3 g((LDW[i] + THREADS - 1) / THREADS, Nd[i]);
    transpose_cast_kernel<<<g, THREADS, 0, stream>>>(Wf[i], Wt[i], Kd[i], Nd[i], LDW[i]);
  }

  float* z_out  = (float*)d_out;                       // [N, 16]
  float* xr_out = (float*)d_out + (size_t)N * LATENT;  // [N, 3703]

  // ---------------- encoder: GEMM-first (dims shrink) ----------------
  const float* H = x;
  int lda = IN_DIM;
  for (int i = 0; i < 4; ++i) {
    int K = Kd[i], No = Nd[i];
    launch_gemm(H, Wt[i], nullptr, bufA, N, K, No, lda, LDW[i], 0, stream);
    size_t tot = (size_t)N * No;
    zero_kernel<<<blocks1d(tot), THREADS, 0, stream>>>(bufB, tot);
    int lf4 = ilog2(No >> 2);
    spmm_kernel<<<blocks1d((size_t)E << lf4), THREADS, 0, stream>>>(
        erow, ecol, eval_, bufA, bufB, E, No, lf4);
    float* outp = (i == 3) ? z_out : bufB;             // layer 3 -> z, no ReLU
    bias_act_kernel<<<blocks1d(tot), THREADS, 0, stream>>>(
        bufB, Bf[i], outp, tot, No - 1, (i < 3) ? 1 : 0);
    H = bufB;
    lda = No;
  }

  // ------------- decoder: SpMM-first, using A(hW) == (Ah)W -------------
  const float* Hd = z_out;
  for (int i = 0; i < 4; ++i) {
    int F = Kd[4 + i], No = Nd[4 + i];
    size_t totF = (size_t)N * F;
    zero_kernel<<<blocks1d(totF), THREADS, 0, stream>>>(bufA, totF);
    int lf4 = ilog2(F >> 2);
    spmm_kernel<<<blocks1d((size_t)E << lf4), THREADS, 0, stream>>>(
        erow, ecol, eval_, Hd, bufA, E, F, lf4);
    float* outp = (i == 3) ? xr_out : bufB;
    launch_gemm(bufA, Wt[4 + i], Bf[4 + i], outp, N, F, No, F, LDW[4 + i],
                (i < 3) ? 2 : 1, stream);
    Hd = bufB;
  }
}
